// FprojNN_StressNN_21921513079370
// MI455X (gfx1250) — matrix operand order repaired and checked by reading, not compiled, not run on hardware
//
#include <hip/hip_runtime.h>
#include <hip/hip_bf16.h>
#include <math.h>

// ---------------------------------------------------------------------------
// MI455X (gfx1250) fused particle-MLP kernel.
//
// Roofline: ~440 GFLOP of GEMM vs ~30MB of HBM I/O -> compute bound.
// v_wmma_f32_16x16x32_bf16 (8x MACs/instr vs f32 16x16x4 WMMA), f32
// accumulation, branch-free tanh-GELU on hardware transcendentals.
// Weights (3.4MB bf16) stay L2-resident.  4 waves/block share one 32-row
// activation tile split along N: 64KB LDS/block -> 5 blocks/WGP -> 20
// waves/WGP for latency hiding, with no increase in L2 weight traffic.
// ---------------------------------------------------------------------------

typedef __attribute__((ext_vector_type(16))) __bf16 v16bf;
typedef __attribute__((ext_vector_type(8)))  float  v8f;

// Cold-path float->bf16 (RNE, integer ops) for pack/prep kernels.
__device__ __forceinline__ unsigned short f2bf(float x) {
  union { float f; unsigned int u; } c; c.f = x;
  unsigned int u = c.u;
  if ((u & 0x7F800000u) == 0x7F800000u) return (unsigned short)(u >> 16); // Inf/NaN
  unsigned int r = u + 0x7FFFu + ((u >> 16) & 1u);                        // RNE
  return (unsigned short)(r >> 16);
}

// Hot-path float->bf16: single hardware v_cvt.
__device__ __forceinline__ unsigned short f2bf_fast(float x) {
  __bf16 b = (__bf16)x;
  return __builtin_bit_cast(unsigned short, b);
}

// Branch-free tanh-approx GELU on hardware trans ops (v_exp_f32 / v_rcp_f32):
// u = 0.79788456x + 0.03567741x^3 ; tanh(u) = 1 - 2/(exp2(2*log2e*u)+1)
__device__ __forceinline__ float fast_gelu(float x) {
  float x2 = x * x;
  float u  = x * fmaf(0.0356774081f, x2, 0.7978845608f);
  float e  = __builtin_amdgcn_exp2f(u * 2.8853900818f);      // exp(2u)
  float t  = 1.0f - 2.0f * __builtin_amdgcn_rcpf(e + 1.0f);  // tanh(u)
  float h  = 0.5f * x;
  return fmaf(h, t, h);
}

// A-matrix (16x32 bf16) per-lane element index inside a K-tile buffer laid out
// as [kt][lane][slot] (512 elems per ktile).  From ISA 7.12.2:
//   lanes 0-15 : M=lane,   slots 0-7 -> K=0..7,   slots 8-15 -> K=16..23
//   lanes16-31 : M=lane-16,slots 0-7 -> K=8..15,  slots 8-15 -> K=24..31
__device__ __forceinline__ int a_index(int m, int k) {
  int kt = k >> 5, kk = k & 31;
  int h, slot;
  if (kk < 16) { h = kk >> 3; slot = kk & 7; }
  else         { int k2 = kk - 16; h = k2 >> 3; slot = 8 + (k2 & 7); }
  return kt * 512 + (m + 16 * h) * 16 + slot;
}

// ---------------------------------------------------------------------------
// Weight repack: f32 [K][N] row-major -> bf16 WMMA B-matrix tiles.
// B tile (32x16, K x N): lane = (n%16) + 16*((kInTile/8)%2), slot = kInTile%8 + 8*(kInTile/16).
// ---------------------------------------------------------------------------
__global__ void pack_w(const float* __restrict__ W, unsigned short* __restrict__ dst,
                       int Kreal, int Nreal, int Ktiles, int Ntiles) {
  int idx = blockIdx.x * blockDim.x + threadIdx.x;
  int total = Ktiles * Ntiles * 512;
  if (idx >= total) return;
  int tile = idx >> 9, e = idx & 511;
  int lane = e >> 4, slot = e & 15;
  int kt = tile / Ntiles, j = tile % Ntiles;
  int kk = 8 * (lane >> 4) + (slot & 7) + 16 * (slot >> 3); // B-matrix mapping
  int n  = j * 16 + (lane & 15);
  int k  = kt * 32 + kk;
  float v = (k < Kreal && n < Nreal) ? W[(size_t)k * Nreal + n] : 0.0f;
  dst[idx] = f2bf(v);
}

// ---------------------------------------------------------------------------
// 3x3 SVD via cyclic Jacobi on A^T A (inputs are near-identity, well cond.)
// ---------------------------------------------------------------------------
__device__ __forceinline__ void jrot(float* M, float* V, int p, int q) {
  float apq = M[p * 3 + q];
  if (fabsf(apq) < 1e-12f) return;
  float app = M[p * 3 + p], aqq = M[q * 3 + q];
  float tau = (aqq - app) / (2.0f * apq);
  float t = ((tau >= 0.0f) ? 1.0f : -1.0f) / (fabsf(tau) + sqrtf(1.0f + tau * tau));
  float c = 1.0f / sqrtf(1.0f + t * t), s = t * c;
  for (int k = 0; k < 3; ++k) {
    float mkp = M[k * 3 + p], mkq = M[k * 3 + q];
    M[k * 3 + p] = c * mkp - s * mkq;
    M[k * 3 + q] = s * mkp + c * mkq;
  }
  for (int k = 0; k < 3; ++k) {
    float mpk = M[p * 3 + k], mqk = M[q * 3 + k];
    M[p * 3 + k] = c * mpk - s * mqk;
    M[q * 3 + k] = s * mpk + c * mqk;
  }
  for (int k = 0; k < 3; ++k) {
    float vkp = V[k * 3 + p], vkq = V[k * 3 + q];
    V[k * 3 + p] = c * vkp - s * vkq;
    V[k * 3 + q] = s * vkp + c * vkq;
  }
}

__device__ void svd3(const float* A, float* U, float* sg, float* Vt) {
  float M[9];
  for (int i = 0; i < 3; ++i)
    for (int j = 0; j < 3; ++j) {
      float acc = 0.0f;
      for (int r = 0; r < 3; ++r) acc += A[r * 3 + i] * A[r * 3 + j];
      M[i * 3 + j] = acc;
    }
  float Vm[9] = {1.f, 0.f, 0.f, 0.f, 1.f, 0.f, 0.f, 0.f, 1.f};
  for (int s = 0; s < 6; ++s) { jrot(M, Vm, 0, 1); jrot(M, Vm, 0, 2); jrot(M, Vm, 1, 2); }
  float ev[3] = {M[0], M[4], M[8]};
  int i0 = 0, i1 = 1, i2 = 2, t;
  if (ev[i0] < ev[i1]) { t = i0; i0 = i1; i1 = t; }
  if (ev[i0] < ev[i2]) { t = i0; i0 = i2; i2 = t; }
  if (ev[i1] < ev[i2]) { t = i1; i1 = i2; i2 = t; }
  int idx[3] = {i0, i1, i2};
  for (int c = 0; c < 3; ++c) {
    sg[c] = sqrtf(fmaxf(ev[idx[c]], 0.0f));
    float v0 = Vm[0 * 3 + idx[c]], v1 = Vm[1 * 3 + idx[c]], v2 = Vm[2 * 3 + idx[c]];
    float w0 = A[0] * v0 + A[1] * v1 + A[2] * v2;
    float w1 = A[3] * v0 + A[4] * v1 + A[5] * v2;
    float w2 = A[6] * v0 + A[7] * v1 + A[8] * v2;
    float nrm = sqrtf(w0 * w0 + w1 * w1 + w2 * w2);
    float inv = (nrm > 1e-12f) ? (1.0f / nrm) : 0.0f;
    U[0 * 3 + c] = w0 * inv; U[1 * 3 + c] = w1 * inv; U[2 * 3 + c] = w2 * inv;
    Vt[c * 3 + 0] = v0; Vt[c * 3 + 1] = v1; Vt[c * 3 + 2] = v2;
  }
}

__device__ __forceinline__ float det3(const float* F) {
  return F[0] * (F[4] * F[8] - F[5] * F[7])
       - F[1] * (F[3] * F[8] - F[5] * F[6])
       + F[2] * (F[3] * F[7] - F[4] * F[6]);
}

// ---------------------------------------------------------------------------
// Per-particle prep: SVDs + feature assembly, written directly in bf16
// A-matrix tile layout: xA[(p/16)*1536 + a_index(p%16, k)].
// ---------------------------------------------------------------------------
__global__ void prep_kernel(const float* __restrict__ Ftmp, const float* __restrict__ F,
                            const float* __restrict__ Cm,   const int* __restrict__ tid,
                            const float* __restrict__ latT,
                            unsigned short* __restrict__ xAf,
                            unsigned short* __restrict__ xAs,
                            float* __restrict__ Rws, int nP) {
  int p = blockIdx.x * blockDim.x + threadIdx.x;
  if (p >= nP) return;
  const int m = p & 15;
  unsigned short* bf = xAf + (size_t)(p >> 4) * 1536;
  unsigned short* bs = xAs + (size_t)(p >> 4) * 1536;
  const float* lat = latT + (size_t)tid[p] * 64;

  // ---- fproj branch features: [Ftmp(9), U(9), sig(3), Vh(9), lat(64)] ----
  float A[9], U[9], sg[3], Vt[9];
  for (int i = 0; i < 9; ++i) A[i] = Ftmp[(size_t)p * 9 + i];
  svd3(A, U, sg, Vt);
  int k = 0;
  for (int i = 0; i < 9; ++i) bf[a_index(m, k++)] = f2bf(A[i]);
  for (int i = 0; i < 9; ++i) bf[a_index(m, k++)] = f2bf(U[i]);
  for (int i = 0; i < 3; ++i) bf[a_index(m, k++)] = f2bf(sg[i]);
  for (int i = 0; i < 9; ++i) bf[a_index(m, k++)] = f2bf(Vt[i]);
  for (int i = 0; i < 64; ++i) bf[a_index(m, k++)] = f2bf(lat[i]);
  while (k < 96) bf[a_index(m, k++)] = 0;

  // ---- stress branch features: [sig2(3), FtF(9), J, lnJ, J1, lnJ1, C(9), lat(64)]
  float Fm[9], U2[9], sg2[3], Vt2[9];
  for (int i = 0; i < 9; ++i) Fm[i] = F[(size_t)p * 9 + i];
  svd3(Fm, U2, sg2, Vt2);
  // R = U2 @ V2  (V2[t][j] = Vt2[j*3+t])
  for (int i = 0; i < 3; ++i)
    for (int j = 0; j < 3; ++j) {
      float acc = 0.0f;
      for (int tt = 0; tt < 3; ++tt) acc += U2[i * 3 + tt] * Vt2[j * 3 + tt];
      Rws[(size_t)p * 9 + i * 3 + j] = acc;
    }
  float J  = fmaxf(det3(Fm), 1e-6f);
  float J1 = fmaxf(Fm[0], 1e-6f);
  k = 0;
  for (int i = 0; i < 3; ++i) bs[a_index(m, k++)] = f2bf(sg2[i]);
  for (int i = 0; i < 3; ++i)
    for (int j = 0; j < 3; ++j) {
      float acc = 0.0f;
      for (int r = 0; r < 3; ++r) acc += Fm[r * 3 + i] * Fm[r * 3 + j];
      bs[a_index(m, k++)] = f2bf(acc);
    }
  bs[a_index(m, k++)] = f2bf(J);
  bs[a_index(m, k++)] = f2bf(logf(J));
  bs[a_index(m, k++)] = f2bf(J1);
  bs[a_index(m, k++)] = f2bf(logf(J1));
  for (int i = 0; i < 9; ++i)  bs[a_index(m, k++)] = f2bf(Cm[(size_t)p * 9 + i]);
  for (int i = 0; i < 64; ++i) bs[a_index(m, k++)] = f2bf(lat[i]);
  while (k < 96) bs[a_index(m, k++)] = 0;
}

// ---------------------------------------------------------------------------
// Fused 5-layer MLP.  One block = 32 particles, 4 waves.  All waves share the
// two 32-row activation buffers (A-tile layout, 2 x 32KB LDS) and split the
// output N dimension (8 of 32 N-tiles per wave per layer), so weights are
// read once per 32 rows while 20 waves/WGP stay resident.
// mode 0: fproj (+ Ftmp residual into d_out).  mode 1: stress (raw 9 outputs
// -> ws, epilogue kernel follows).
// ---------------------------------------------------------------------------
__global__ __launch_bounds__(128) void mlp_fused(
    const unsigned short* __restrict__ xA,
    const unsigned short* __restrict__ Wp,
    const float* __restrict__ b0, const float* __restrict__ b1,
    const float* __restrict__ b2, const float* __restrict__ b3,
    const float* __restrict__ b4,
    const float* __restrict__ resid,
    float* __restrict__ outp, int mode) {
  __shared__ __attribute__((aligned(32))) unsigned short act[2][16384]; // 64KB
  const int wave = threadIdx.x >> 5;
  const int lane = threadIdx.x & 31;
  const int g = blockIdx.x;
  const int rowbase = g << 5;
  const int hi = lane >> 4;
  const int nlo = lane & 15;
  // Epilogue A-layout store addressing (hand-simplified a_index):
  //   ai(j, v) = (j>>1)*512 + (mrow + 16*(nlo>>3))*16 + (nlo&7) + 8*(j&1)
  const int lane2base = 16 * (nlo >> 3) * 16 + (nlo & 7); // (16*h)*16 + slot0

  { // stage both 16-row input tiles as uint4 (3072 ushorts = 384 x 16B)
    uint4* a4 = (uint4*)&act[0][0];
    const uint4* s4 = (const uint4*)(xA + (size_t)g * 3072);
    for (int i = threadIdx.x; i < 384; i += 128) {
      int mt = (i >= 192) ? 1 : 0;
      a4[mt * 1024 + (i - mt * 192)] = s4[i];   // mtile1 region starts at uint4 #1024
    }
  }
  __syncthreads();

  auto layer = [&](int inBuf, int outBuf, const unsigned short* wb,
                   const float* bias, int Kt, int Nt) {
    const int per = Nt >> 2;              // N-tiles per wave
    const int jend = (wave + 1) * per;
    for (int j = wave * per; j < jend; ++j) {
      v8f acc0 = {}; v8f acc1 = {};
      for (int kt = 0; kt < Kt; ++kt) {
        v16bf bm = *(const v16bf*)(wb + (((size_t)kt * Nt + j) << 9) + (lane << 4));
        v16bf a0 = *(const v16bf*)(&act[inBuf][kt * 512 + (lane << 4)]);
        v16bf a1 = *(const v16bf*)(&act[inBuf][8192 + kt * 512 + (lane << 4)]);
        acc0 = __builtin_amdgcn_wmma_f32_16x16x32_bf16(false, a0, false, bm,
                                                       (short)0, acc0, false, false);
        acc1 = __builtin_amdgcn_wmma_f32_16x16x32_bf16(false, a1, false, bm,
                                                       (short)0, acc1, false, false);
      }
      float bv = bias[(j << 4) + nlo];
      int aibase = (j >> 1) * 512 + lane2base + ((j & 1) << 3);
      for (int v = 0; v < 8; ++v) {
        int mrow = v + 8 * hi;                 // D layout: VGPR v -> row v / 8+v
        int ai = aibase + mrow * 16;
        act[outBuf][ai]        = f2bf_fast(fast_gelu(acc0[v] + bv));
        act[outBuf][8192 + ai] = f2bf_fast(fast_gelu(acc1[v] + bv));
      }
    }
    __syncthreads();
  };

  layer(0, 1, Wp + 0,      b0, 3,  32);
  layer(1, 0, Wp + 49152,  b1, 16, 32);
  layer(0, 1, Wp + 311296, b2, 16, 32);
  layer(1, 0, Wp + 573440, b3, 16, 32);

  if (wave == 0) { // final layer: K=512, N=16 (cols 0..8 valid), fused epilogue
    const unsigned short* wb = Wp + 835584;
    v8f acc0 = {}; v8f acc1 = {};
    for (int kt = 0; kt < 16; ++kt) {
      v16bf bm = *(const v16bf*)(wb + ((size_t)kt << 9) + (lane << 4));
      v16bf a0 = *(const v16bf*)(&act[0][kt * 512 + (lane << 4)]);
      v16bf a1 = *(const v16bf*)(&act[0][8192 + kt * 512 + (lane << 4)]);
      acc0 = __builtin_amdgcn_wmma_f32_16x16x32_bf16(false, a0, false, bm,
                                                     (short)0, acc0, false, false);
      acc1 = __builtin_amdgcn_wmma_f32_16x16x32_bf16(false, a1, false, bm,
                                                     (short)0, acc1, false, false);
    }
    if (nlo < 9) {
      float bv = b4[nlo];
      for (int v = 0; v < 8; ++v) {
        int r0 = rowbase + v + 8 * hi;
        int r1 = r0 + 16;
        float y0 = acc0[v] + bv;
        float y1 = acc1[v] + bv;
        if (mode == 0) {
          outp[(size_t)r0 * 9 + nlo] = y0 + resid[(size_t)r0 * 9 + nlo];
          outp[(size_t)r1 * 9 + nlo] = y1 + resid[(size_t)r1 * 9 + nlo];
        } else {
          outp[((size_t)r0 << 4) + nlo] = y0;
          outp[((size_t)r1 << 4) + nlo] = y1;
        }
      }
    }
  }
}

// ---------------------------------------------------------------------------
// Stress epilogue: S = 0.5(S + S^T); out = R @ S  (second half of d_out)
// ---------------------------------------------------------------------------
__global__ void stress_epi(const float* __restrict__ o16, const float* __restrict__ Rws,
                           float* __restrict__ out, int nP) {
  int p = blockIdx.x * blockDim.x + threadIdx.x;
  if (p >= nP) return;
  float S[9], Sy[9], R[9];
  for (int i = 0; i < 9; ++i) S[i] = o16[((size_t)p << 4) + i];
  for (int i = 0; i < 3; ++i)
    for (int j = 0; j < 3; ++j)
      Sy[i * 3 + j] = 0.5f * (S[i * 3 + j] + S[j * 3 + i]);
  for (int i = 0; i < 9; ++i) R[i] = Rws[(size_t)p * 9 + i];
  float* dst = out + (size_t)nP * 9 + (size_t)p * 9;
  for (int i = 0; i < 3; ++i)
    for (int kk = 0; kk < 3; ++kk) {
      float acc = 0.0f;
      for (int j = 0; j < 3; ++j) acc += R[i * 3 + j] * Sy[j * 3 + kk];
      dst[i * 3 + kk] = acc;
    }
}

// ---------------------------------------------------------------------------
extern "C" void kernel_launch(void* const* d_in, const int* in_sizes, int n_in,
                              void* d_out, int out_size, void* d_ws, size_t ws_size,
                              hipStream_t stream) {
  (void)n_in; (void)out_size; (void)ws_size;
  const float* Ftmp = (const float*)d_in[0];
  const float* F    = (const float*)d_in[1];
  const float* Cm   = (const float*)d_in[2];
  const int*   tid  = (const int*)d_in[3];
  const float* lat  = (const float*)d_in[4];
  const float* fW[5]; const float* fB[5]; const float* sW[5]; const float* sB[5];
  for (int i = 0; i < 5; ++i) {
    fW[i] = (const float*)d_in[5 + i];
    fB[i] = (const float*)d_in[10 + i];
    sW[i] = (const float*)d_in[15 + i];
    sB[i] = (const float*)d_in[20 + i];
  }
  const int nP = in_sizes[0] / 9;   // 131072
  const int nW16 = nP / 16;

  // scratch carve-out (~67MB total)
  char* ws = (char*)d_ws; size_t off = 0;
  auto carve = [&](size_t bytes) -> void* {
    void* p = ws + off; off = (off + bytes + 255) & ~(size_t)255; return p;
  };
  const size_t WELEMS = 843776;     // packed bf16 elems per network
  unsigned short* WfP = (unsigned short*)carve(WELEMS * 2);
  unsigned short* WsP = (unsigned short*)carve(WELEMS * 2);
  unsigned short* xAf = (unsigned short*)carve((size_t)nW16 * 1536 * 2);
  unsigned short* xAs = (unsigned short*)carve((size_t)nW16 * 1536 * 2);
  float* Rws = (float*)carve((size_t)nP * 9 * 4);
  float* o16 = (float*)carve((size_t)nP * 16 * 4);

  static const int    Kt[5]   = {3, 16, 16, 16, 16};
  static const int    Nt[5]   = {32, 32, 32, 32, 1};
  static const size_t loff[5] = {0, 49152, 311296, 573440, 835584};
  static const int    KrF[5]  = {94, 512, 512, 512, 512};
  static const int    KrS[5]  = {89, 512, 512, 512, 512};
  static const int    Nr[5]   = {512, 512, 512, 512, 9};

  for (int l = 0; l < 5; ++l) {
    int tot = Kt[l] * Nt[l] * 512;
    pack_w<<<(tot + 255) / 256, 256, 0, stream>>>(fW[l], WfP + loff[l], KrF[l], Nr[l], Kt[l], Nt[l]);
    pack_w<<<(tot + 255) / 256, 256, 0, stream>>>(sW[l], WsP + loff[l], KrS[l], Nr[l], Kt[l], Nt[l]);
  }
  prep_kernel<<<(nP + 255) / 256, 256, 0, stream>>>(Ftmp, F, Cm, tid, lat, xAf, xAs, Rws, nP);
  mlp_fused<<<nP / 32, 128, 0, stream>>>(xAf, WfP, fB[0], fB[1], fB[2], fB[3], fB[4],
                                         Ftmp, (float*)d_out, 0);
  mlp_fused<<<nP / 32, 128, 0, stream>>>(xAs, WsP, sB[0], sB[1], sB[2], sB[3], sB[4],
                                         nullptr, o16, 1);
  stress_epi<<<(nP + 255) / 256, 256, 0, stream>>>(o16, Rws, (float*)d_out, nP);
}
